// Cross_Attention_76029511073874
// MI455X (gfx1250) — matrix operand order repaired and checked by reading, not compiled, not run on hardware
//
#include <hip/hip_runtime.h>

// Cross-attention, CDNA5 (gfx1250), wave32 WMMA bf16 flash-attention.
// B=4, Sx=Sy=4096, D=1024, H=64.
//
// Workspace layout (bytes), total ~6.5 MB:
//   [0, 384K)            : WqT/WkT/WvT  bf16, each [64][1024] (transposed)
//   [512K, 512K+2M)      : q  bf16 [B][S][64] row-major
//   [512K+2M, 512K+4M)   : k  bf16 [B][S][64] row-major
//   [512K+4M, 512K+6M)   : vT bf16 [B][64][S] (transposed for PV B-fragments)

#define S_LEN 4096
#define D_DIM 1024
#define H_DIM 64
#define B_DIM 4

typedef __attribute__((ext_vector_type(16))) __bf16 v16bf;
typedef __attribute__((ext_vector_type(8)))  __bf16 v8bf;
typedef __attribute__((ext_vector_type(8)))  float  v8f;

static __device__ __forceinline__ v8f wmma_bf16(v16bf a, v16bf b, v8f c) {
  // D = A(16x32) * B(32x16) + C, f32 accumulate
  return __builtin_amdgcn_wmma_f32_16x16x32_bf16(
      /*neg_a=*/false, a, /*neg_b=*/false, b,
      /*c_mod=*/(short)0, c, /*reuse_a=*/false, /*reuse_b=*/false);
}

// ---------------------------------------------------------------------------
// Kernel 0: convert Wq/Wk/Wv [1024][64] f32 -> transposed bf16 [64][1024]
// ---------------------------------------------------------------------------
__global__ __launch_bounds__(256) void wt_convert_kernel(
    const float* __restrict__ Wq, const float* __restrict__ Wk,
    const float* __restrict__ Wv, __bf16* __restrict__ wt) {
  int id = blockIdx.x * 256 + threadIdx.x;       // 3 * 1024 * 64 total
  int n = id & 63;
  int k = (id >> 6) & 1023;
  int w = id >> 16;
  const float* src = (w == 0) ? Wq : (w == 1) ? Wk : Wv;
  wt[(size_t)w * 65536 + (size_t)n * 1024 + k] = (__bf16)src[k * 64 + n];
}

// ---------------------------------------------------------------------------
// Kernel 1: projections. which: 0 -> q = y@Wq (row-major out)
//                               1 -> k = x@Wk (row-major out)
//                               2 -> v = x@Wv (TRANSPOSED out [64][S])
// Block = 256 thr = 8 waves, wave owns a 16x64 output tile, K=1024.
// Two-phase ping-pong pipeline (stride 64 in K): phase X's WMMAs overlap
// phase Y's loads; statically distinct buffers -> no register-rotation movs.
// ---------------------------------------------------------------------------
__global__ __launch_bounds__(256) void proj_kernel(
    const float* __restrict__ x, const float* __restrict__ y,
    const __bf16* __restrict__ wt,
    __bf16* __restrict__ qbuf, __bf16* __restrict__ kbuf,
    __bf16* __restrict__ vtbuf) {
  const int which = blockIdx.z;
  const int b     = blockIdx.y;
  const int wave  = threadIdx.x >> 5;
  const int lane  = threadIdx.x & 31;
  const int m     = lane & 15;         // M row (A) / N col (B,C,D)
  const int hi    = lane >> 4;         // lane half select
  const int row0  = blockIdx.x * 128 + wave * 16;

  const float*  src = (which == 0) ? y : x;
  const __bf16* W   = wt + (size_t)which * 65536;   // [64][1024] transposed

  const v8f vz = {0.f,0.f,0.f,0.f,0.f,0.f,0.f,0.f};
  v8f acc[4] = {vz, vz, vz, vz};

  const float*  arow  = src + ((size_t)b * S_LEN + row0 + m) * D_DIM;
  const __bf16* wbase = W + 16 * hi;   // + (nt*16+m)*1024 + k0

  // ---- prologue: phase-A fragments for k0 = 0 ----
  v8f   fa0 = *(const v8f*)(arow + 16 * hi);
  v8f   fa1 = *(const v8f*)(arow + 8 + 16 * hi);
  v16bf ba[4];
#pragma unroll
  for (int nt = 0; nt < 4; ++nt)
    ba[nt] = *(const v16bf*)(wbase + (size_t)(nt * 16 + m) * 1024);

  for (int k0 = 0; k0 < D_DIM; k0 += 64) {
    // load phase-B fragments (K chunk k0+32; always in range, D % 64 == 0)
    v8f   fb0 = *(const v8f*)(arow + k0 + 32 + 16 * hi);
    v8f   fb1 = *(const v8f*)(arow + k0 + 40 + 16 * hi);
    v16bf bb[4];
#pragma unroll
    for (int nt = 0; nt < 4; ++nt)
      bb[nt] = *(const v16bf*)(wbase + (size_t)(nt * 16 + m) * 1024 + k0 + 32);

    // compute phase A  (A frag: elems 0..15 = K k0+16*hi..)
    v16bf a;
#pragma unroll
    for (int e = 0; e < 8; ++e) { a[e] = (__bf16)fa0[e]; a[e + 8] = (__bf16)fa1[e]; }
#pragma unroll
    for (int nt = 0; nt < 4; ++nt) acc[nt] = wmma_bf16(a, ba[nt], acc[nt]);

    // load phase-A fragments for next stride (K chunk k0+64)
    if (k0 + 64 < D_DIM) {
      fa0 = *(const v8f*)(arow + k0 + 64 + 16 * hi);
      fa1 = *(const v8f*)(arow + k0 + 72 + 16 * hi);
#pragma unroll
      for (int nt = 0; nt < 4; ++nt)
        ba[nt] = *(const v16bf*)(wbase + (size_t)(nt * 16 + m) * 1024 + k0 + 64);
    }

    // compute phase B
    v16bf a2;
#pragma unroll
    for (int e = 0; e < 8; ++e) { a2[e] = (__bf16)fb0[e]; a2[e + 8] = (__bf16)fb1[e]; }
#pragma unroll
    for (int nt = 0; nt < 4; ++nt) acc[nt] = wmma_bf16(a2, bb[nt], acc[nt]);
  }

  // C/D layout: lane col = m, VGPR r -> row (row0 + r + 8*hi)
  if (which == 2) {
#pragma unroll
    for (int nt = 0; nt < 4; ++nt) {
      v8bf pk;
#pragma unroll
      for (int r = 0; r < 8; ++r) pk[r] = (__bf16)acc[nt][r];
      *(v8bf*)(vtbuf + ((size_t)b * H_DIM + nt * 16 + m) * S_LEN + row0 + 8 * hi) = pk;
    }
  } else {
    __bf16* dst = (which == 0) ? qbuf : kbuf;
#pragma unroll
    for (int nt = 0; nt < 4; ++nt)
#pragma unroll
      for (int r = 0; r < 8; ++r)
        dst[((size_t)b * S_LEN + row0 + r + 8 * hi) * H_DIM + nt * 16 + m] =
            (__bf16)acc[nt][r];
  }
}

// ---------------------------------------------------------------------------
// Kernel 2: flash attention. Block = 128 thr = 4 waves; wave owns 16 q rows.
// Keys streamed in blocks of 32 (2 S-tiles -> one K=32 PV WMMA). V fragments
// are loaded BEFORE the LDS reshape barriers so they overlap the softmax VALU.
// ---------------------------------------------------------------------------
__global__ __launch_bounds__(128) void attn_kernel(
    const __bf16* __restrict__ qb, const __bf16* __restrict__ kb,
    const __bf16* __restrict__ vtb,
    const int* __restrict__ vlx_p, const int* __restrict__ vly_p,
    const int* __restrict__ causal_p, float* __restrict__ out) {
  const int b    = blockIdx.y;
  const int wave = threadIdx.x >> 5;
  const int lane = threadIdx.x & 31;
  const int m    = lane & 15;
  const int hi   = lane >> 4;
  const int qr0  = blockIdx.x * 64 + wave * 16;

  const int causal = causal_p[0];
  const int vlx    = vlx_p[b];
  const int vly    = vly_p[b];
  const float NEG_INF = -__builtin_inff();

  // Per-wave LDS tile for P C-layout -> A-layout reshape. Rows padded to 40
  // halves (80 B) so the 16 b128 reads land on distinct bank groups.
  __shared__ __align__(32) __bf16 ptile[4][16 * 40];
  __bf16* myp = ptile[wave];

  // Load Q A-fragments (K=64 -> 2 chunks of 32) once.
  const __bf16* qrow = qb + ((size_t)b * S_LEN + qr0 + m) * H_DIM;
  v16bf qa[2];
#pragma unroll
  for (int c = 0; c < 2; ++c) {
    v8bf lo  = *(const v8bf*)(qrow + c * 32 + 16 * hi);
    v8bf hi8 = *(const v8bf*)(qrow + c * 32 + 8 + 16 * hi);
#pragma unroll
    for (int e = 0; e < 8; ++e) { qa[c][e] = lo[e]; qa[c][e + 8] = hi8[e]; }
  }

  const float scale = 0.125f;  // 64^-0.5
  float mrow[8], lrow[8];
#pragma unroll
  for (int r = 0; r < 8; ++r) { mrow[r] = NEG_INF; lrow[r] = 0.f; }
  const v8f vz = {0.f,0.f,0.f,0.f,0.f,0.f,0.f,0.f};
  v8f oacc[4] = {vz, vz, vz, vz};

  int kend = vlx;                                   // key mask bound
  if (causal) { int ce = qr0 + 16; if (ce < kend) kend = ce; }
  if (qr0 >= vly) kend = 0;                         // whole wave's rows masked

  for (int j = 0; j < kend; j += 32) {
    if (j + 32 < kend) {                            // prefetch next K tile
      __builtin_prefetch(kb + ((size_t)b * S_LEN + j + 32 + m) * H_DIM, 0, 1);
      __builtin_prefetch(kb + ((size_t)b * S_LEN + j + 48 + m) * H_DIM, 0, 1);
    }
    // --- S = Q K^T for two 16-key tiles, then scale + mask (C layout) ---
    float sv[2][8];
#pragma unroll
    for (int t = 0; t < 2; ++t) {
      v8f sacc = vz;
#pragma unroll
      for (int c = 0; c < 2; ++c) {
        // B frag: lane col = key (j+16t+m); elems = feat c*32+16*hi .. +15
        v16bf kf = *(const v16bf*)(kb + ((size_t)b * S_LEN + j + t * 16 + m) * H_DIM
                                   + c * 32 + 16 * hi);
        sacc = wmma_bf16(qa[c], kf, sacc);
      }
      const int key = j + t * 16 + m;
      const bool kvalid = (key < vlx);
#pragma unroll
      for (int r = 0; r < 8; ++r) {
        const int qr = qr0 + r + 8 * hi;
        bool ok = kvalid && (qr < vly) && (!causal || key <= qr);
        sv[t][r] = ok ? sacc[r] * scale : NEG_INF;
      }
    }
    // --- V B-fragments issued early: overlap softmax VALU + LDS reshape ---
    v16bf vf[4];
#pragma unroll
    for (int nt = 0; nt < 4; ++nt)
      vf[nt] = *(const v16bf*)(vtb + ((size_t)b * H_DIM + nt * 16 + m) * S_LEN
                               + j + 16 * hi);

    // --- online softmax: row max across the 16 lanes of each half ---
    float tmx[8];
#pragma unroll
    for (int r = 0; r < 8; ++r) tmx[r] = fmaxf(sv[0][r], sv[1][r]);
#pragma unroll
    for (int off = 1; off < 16; off <<= 1)
#pragma unroll
      for (int r = 0; r < 8; ++r)
        tmx[r] = fmaxf(tmx[r], __shfl_xor(tmx[r], off, 32));
    float corr[8];
#pragma unroll
    for (int r = 0; r < 8; ++r) {
      float mn = fmaxf(mrow[r], tmx[r]);
      corr[r] = (mn == NEG_INF) ? 1.f : __expf(mrow[r] - mn);
      mrow[r] = mn;
      lrow[r] *= corr[r];
    }
    // --- P = exp(S - m), write to LDS (C layout), accumulate row sums ---
    float rsum[8];
#pragma unroll
    for (int r = 0; r < 8; ++r) rsum[r] = 0.f;
#pragma unroll
    for (int t = 0; t < 2; ++t)
#pragma unroll
      for (int r = 0; r < 8; ++r) {
        float p = (sv[t][r] == NEG_INF) ? 0.f : __expf(sv[t][r] - mrow[r]);
        rsum[r] += p;
        myp[(r + 8 * hi) * 40 + t * 16 + m] = (__bf16)p;
      }
#pragma unroll
    for (int off = 1; off < 16; off <<= 1)
#pragma unroll
      for (int r = 0; r < 8; ++r)
        rsum[r] += __shfl_xor(rsum[r], off, 32);
#pragma unroll
    for (int r = 0; r < 8; ++r) lrow[r] += rsum[r];
#pragma unroll
    for (int nt = 0; nt < 4; ++nt)
#pragma unroll
      for (int r = 0; r < 8; ++r) oacc[nt][r] *= corr[r];

    // --- reshape P: LDS round-trip into A-fragment layout (K=32 keys) ---
    __builtin_amdgcn_wave_barrier();   // no reordering; same-wave DS is in-order
    v16bf pa;
    {
      v8bf lo  = *(const v8bf*)(myp + m * 40 + 16 * hi);
      v8bf hi8 = *(const v8bf*)(myp + m * 40 + 8 + 16 * hi);
#pragma unroll
      for (int e = 0; e < 8; ++e) { pa[e] = lo[e]; pa[e + 8] = hi8[e]; }
    }
    __builtin_amdgcn_wave_barrier();   // keep next iter's stores after the reads

    // --- O += P V (fragments already resident) ---
#pragma unroll
    for (int nt = 0; nt < 4; ++nt) oacc[nt] = wmma_bf16(pa, vf[nt], oacc[nt]);
  }

  // --- epilogue: normalize; fully-masked rows (l==0) -> 0 (NaN-safe) ---
#pragma unroll
  for (int nt = 0; nt < 4; ++nt)
#pragma unroll
    for (int r = 0; r < 8; ++r) {
      float o = (lrow[r] > 0.f) ? oacc[nt][r] / lrow[r] : 0.f;
      out[((size_t)b * S_LEN + qr0 + r + 8 * hi) * H_DIM + nt * 16 + m] = o;
    }
}

// ---------------------------------------------------------------------------
extern "C" void kernel_launch(void* const* d_in, const int* in_sizes, int n_in,
                              void* d_out, int out_size, void* d_ws, size_t ws_size,
                              hipStream_t stream) {
  const float* x  = (const float*)d_in[0];
  const float* y  = (const float*)d_in[1];
  const int* vlx  = (const int*)d_in[2];
  const int* vly  = (const int*)d_in[3];
  const int* caus = (const int*)d_in[4];
  const float* Wq = (const float*)d_in[5];
  const float* Wk = (const float*)d_in[6];
  const float* Wv = (const float*)d_in[7];
  float* out = (float*)d_out;

  char* ws = (char*)d_ws;
  __bf16* wt   = (__bf16*)(ws);                                  // 384 KB
  __bf16* qbuf = (__bf16*)(ws + 512 * 1024);                     // 2 MB
  __bf16* kbuf = (__bf16*)(ws + 512 * 1024 + 2 * 1024 * 1024);   // 2 MB
  __bf16* vtb  = (__bf16*)(ws + 512 * 1024 + 4 * 1024 * 1024);   // 2 MB

  wt_convert_kernel<<<768, 256, 0, stream>>>(Wq, Wk, Wv, wt);
  proj_kernel<<<dim3(S_LEN / 128, B_DIM, 3), 256, 0, stream>>>(x, y, wt, qbuf, kbuf, vtb);
  attn_kernel<<<dim3(S_LEN / 64, B_DIM), 128, 0, stream>>>(qbuf, kbuf, vtb, vlx, vly, caus, out);
}
